// AttentionDecoder_73065983639983
// MI455X (gfx1250) — compile-verified, matrix operand order checked by reading, not run on hardware
//
#include <hip/hip_runtime.h>
#include <hip/hip_bf16.h>

// Problem constants
#define CB 32      // batch
#define CS 1024    // source length
#define CE 512     // encoder dim
#define CD 512     // decoder dim
#define CA 128     // attention dim
#define CV 32000   // vocab
#define CT 64      // target length
#define CG 2048    // 4*D gates
#define CDE 1024   // D+E

typedef __bf16 bf16;
typedef __attribute__((ext_vector_type(16))) __bf16 v16bf;
typedef __attribute__((ext_vector_type(8)))  float  v8f;
typedef __attribute__((ext_vector_type(4)))  unsigned int u32x4;

// ---------------------------------------------------------------------------
// WMMA helpers (gfx1250 wave32). 16-bit A 16x32 layout (ISA 7.12.2):
// lanes 0-15 hold row M=lane, K in {0..7,16..23}; lanes 16-31 hold the same
// row's K in {8..15,24..31}. B (=W row-major [N,K], i.e. Kx16 col-major tile)
// has the mirrored layout, so one loader serves A and B fragments.
// ---------------------------------------------------------------------------
__device__ __forceinline__ v16bf ldfrag(const bf16* __restrict__ tile, int ld) {
  int lane = threadIdx.x & 31;
  const bf16* p = tile + (size_t)(lane & 15) * ld + ((lane >> 4) << 3);
  union { u32x4 u[2]; v16bf v; } w;
  w.u[0] = *(const u32x4*)p;
  w.u[1] = *(const u32x4*)(p + 16);
  return w.v;
}

__device__ __forceinline__ v8f wmma_bf16(v16bf a, v16bf b, v8f c) {
  return __builtin_amdgcn_wmma_f32_16x16x32_bf16(false, a, false, b, (short)0, c,
                                                 false, false);
}

// ---------------------------------------------------------------------------
// fp32 -> bf16 conversion (grid-stride)
// ---------------------------------------------------------------------------
__global__ void f2bf(const float* __restrict__ src, bf16* __restrict__ dst,
                     long long n) {
  long long i = (long long)blockIdx.x * blockDim.x + threadIdx.x;
  long long stride = (long long)gridDim.x * blockDim.x;
  for (; i < n; i += stride) dst[i] = (bf16)src[i];
}

__global__ void zero_f32(float* __restrict__ p, int n) {
  int i = blockIdx.x * blockDim.x + threadIdx.x;
  if (i < n) p[i] = 0.f;
}
__global__ void zero_bf(bf16* __restrict__ p, int n) {
  int i = blockIdx.x * blockDim.x + threadIdx.x;
  if (i < n) p[i] = (bf16)0.f;
}

// ---------------------------------------------------------------------------
// We = enc(bf16) [32768,512] @ W_a(bf16)^T [512,128] + b_Wa  -> f32 [32768,128]
// one wave per 16x16 tile; 2048 m-tiles x 8 n-tiles
// ---------------------------------------------------------------------------
__global__ void we_gemm(const bf16* __restrict__ Abf, const bf16* __restrict__ Bbf,
                        const float* __restrict__ bias, float* __restrict__ Dst) {
  int wave = (blockIdx.x * blockDim.x + threadIdx.x) >> 5;
  int mt = wave >> 3;
  int nt = wave & 7;
  const bf16* Ab = Abf + (size_t)mt * 16 * CE;
  const bf16* Bb = Bbf + (size_t)nt * 16 * CE;
  v8f c = {};
  for (int k = 0; k < CE; k += 32)
    c = wmma_bf16(ldfrag(Ab + k, CE), ldfrag(Bb + k, CE), c);
  int lane = threadIdx.x & 31;
  int n = lane & 15, mb = (lane >> 4) << 3;
  float bv = bias[nt * 16 + n];
  float* drow = Dst + ((size_t)mt * 16 + mb) * CA + nt * 16 + n;
#pragma unroll
  for (int i = 0; i < 8; ++i) drow[(size_t)i * CA] = c[i] + bv;
}

// ---------------------------------------------------------------------------
// gates[32,2048] = Xa[32,Ka] @ Wa^T + Xb[32,Kb] @ Wb^T   (weights [2048,K] row-major)
// 256 waves: 128 n-tiles x 2 m-tiles
// ---------------------------------------------------------------------------
__global__ void gates_gemm(const bf16* __restrict__ Xa, int lda, int Ka,
                           const bf16* __restrict__ Wa_,
                           const bf16* __restrict__ Xb, int ldb, int Kb,
                           const bf16* __restrict__ Wb_,
                           float* __restrict__ Dst) {
  int wave = (blockIdx.x * blockDim.x + threadIdx.x) >> 5;
  int nt = wave >> 1;
  int mt = wave & 1;
  v8f c = {};
  const bf16* Ar = Xa + (size_t)mt * 16 * lda;
  const bf16* Wr = Wa_ + (size_t)nt * 16 * Ka;
  for (int k = 0; k < Ka; k += 32)
    c = wmma_bf16(ldfrag(Ar + k, lda), ldfrag(Wr + k, Ka), c);
  const bf16* Br = Xb + (size_t)mt * 16 * ldb;
  const bf16* W2 = Wb_ + (size_t)nt * 16 * Kb;
  for (int k = 0; k < Kb; k += 32)
    c = wmma_bf16(ldfrag(Br + k, ldb), ldfrag(W2 + k, Kb), c);
  int lane = threadIdx.x & 31;
  int n = lane & 15, mb = (lane >> 4) << 3;
  float* drow = Dst + ((size_t)mt * 16 + mb) * CG + nt * 16 + n;
#pragma unroll
  for (int i = 0; i < 8; ++i) drow[(size_t)i * CG] = c[i];
}

// ---------------------------------------------------------------------------
// out[b, t, :] = Y[32,1024] @ W_out^T [1024,32000] + b_out   (PyTorch i,f,g,o order upstream)
// 500 waves, each: 64 output cols (4 n-tiles) x both 16-row m-tiles; A frags reused 4x
// ---------------------------------------------------------------------------
__global__ void out_gemm(const bf16* __restrict__ Y, const bf16* __restrict__ Wo,
                         const float* __restrict__ bias, float* __restrict__ Out,
                         int t) {
  int wave = (blockIdx.x * blockDim.x + threadIdx.x) >> 5;
  int n0 = wave * 64;
  v8f c0[4] = {}, c1[4] = {};
  for (int k = 0; k < CDE; k += 32) {
    v16bf a0 = ldfrag(Y + k, CDE);
    v16bf a1 = ldfrag(Y + (size_t)16 * CDE + k, CDE);
#pragma unroll
    for (int j = 0; j < 4; ++j) {
      v16bf b = ldfrag(Wo + (size_t)(n0 + j * 16) * CDE + k, CDE);
      c0[j] = wmma_bf16(a0, b, c0[j]);
      c1[j] = wmma_bf16(a1, b, c1[j]);
    }
  }
  int lane = threadIdx.x & 31;
  int n = lane & 15, mb = (lane >> 4) << 3;
#pragma unroll
  for (int j = 0; j < 4; ++j) {
    int col = n0 + j * 16 + n;
    float bv = bias[col];
#pragma unroll
    for (int i = 0; i < 8; ++i) {
      int m0 = mb + i;
      Out[((size_t)m0 * (CT - 1) + t) * CV + col] = c0[j][i] + bv;
      Out[((size_t)(m0 + 16) * (CT - 1) + t) * CV + col] = c1[j][i] + bv;
    }
  }
}

// ---------------------------------------------------------------------------
// LSTM cell elementwise: gates [32,2048] (+b_ih+b_hh) -> c,h ; h also as bf16
// ---------------------------------------------------------------------------
__global__ void lstm_cell(const float* __restrict__ gates,
                          const float* __restrict__ b_ih,
                          const float* __restrict__ b_hh,
                          float* __restrict__ cst, float* __restrict__ hst,
                          bf16* __restrict__ hbf, int ldh) {
  int idx = blockIdx.x * blockDim.x + threadIdx.x;
  if (idx >= CB * CD) return;
  int b = idx / CD, d = idx % CD;
  const float* g = gates + (size_t)b * CG;
  float gi = g[d]          + b_ih[d]          + b_hh[d];
  float gf = g[d +   CD]   + b_ih[d +   CD]   + b_hh[d +   CD];
  float gg = g[d + 2*CD]   + b_ih[d + 2*CD]   + b_hh[d + 2*CD];
  float go = g[d + 3*CD]   + b_ih[d + 3*CD]   + b_hh[d + 3*CD];
  float si = 1.f / (1.f + __expf(-gi));
  float sf = 1.f / (1.f + __expf(-gf));
  float so = 1.f / (1.f + __expf(-go));
  float cc = sf * cst[idx] + si * tanhf(gg);
  float hh = so * tanhf(cc);
  cst[idx] = cc;
  if (hst) hst[idx] = hh;
  hbf[(size_t)b * ldh + d] = (bf16)hh;
}

// ---------------------------------------------------------------------------
// Attention: q = h1 @ U_a^T + b_Ua ; s = tanh(We + q)·v_a + b_va ; softmax ;
// ctx = aw @ enc (zeroed at t==0); writes x_bf = [emb_tok, ctx] and y ctx-half.
// One block per batch element.
// ---------------------------------------------------------------------------
__global__ void attn_kernel(const float* __restrict__ enc, const float* __restrict__ We,
                            const float* __restrict__ h1, const float* __restrict__ U_a,
                            const float* __restrict__ b_Ua, const float* __restrict__ v_a,
                            const float* __restrict__ b_va, const float* __restrict__ emb,
                            const int* __restrict__ targets, int t,
                            bf16* __restrict__ xbf, bf16* __restrict__ ybf) {
  __shared__ float sh_h1[CD];
  __shared__ float sh_q[CA];
  __shared__ float sh_s[CS];
  __shared__ float red[256];
  int b = blockIdx.x, tid = threadIdx.x;
  for (int i = tid; i < CD; i += 256) sh_h1[i] = h1[(size_t)b * CD + i];
  __syncthreads();
  if (tid < CA) {
    float acc = b_Ua[tid];
    const float* u = U_a + (size_t)tid * CD;
    for (int d = 0; d < CD; ++d) acc += sh_h1[d] * u[d];
    sh_q[tid] = acc;
  }
  __syncthreads();
  for (int s = tid; s < CS; s += 256) {
    const float* w = We + ((size_t)b * CS + s) * CA;
    float acc = b_va[0];
    for (int a = 0; a < CA; ++a) acc += v_a[a] * tanhf(w[a] + sh_q[a]);
    sh_s[s] = acc;
  }
  __syncthreads();
  // softmax over S
  float m = -1e30f;
  for (int s = tid; s < CS; s += 256) m = fmaxf(m, sh_s[s]);
  red[tid] = m; __syncthreads();
  for (int off = 128; off > 0; off >>= 1) {
    if (tid < off) red[tid] = fmaxf(red[tid], red[tid + off]);
    __syncthreads();
  }
  m = red[0]; __syncthreads();
  float sum = 0.f;
  for (int s = tid; s < CS; s += 256) {
    float e = __expf(sh_s[s] - m);
    sh_s[s] = e; sum += e;
  }
  red[tid] = sum; __syncthreads();
  for (int off = 128; off > 0; off >>= 1) {
    if (tid < off) red[tid] += red[tid + off];
    __syncthreads();
  }
  float inv = 1.f / red[0];
  __syncthreads();
  for (int s = tid; s < CS; s += 256) sh_s[s] *= inv;
  __syncthreads();
  // ctx = aw @ enc  (coalesced across lanes in e)
  for (int e = tid; e < CE; e += 256) {
    float acc = 0.f;
    const float* ec = enc + ((size_t)b * CS) * CE + e;
    for (int s = 0; s < CS; ++s) acc += sh_s[s] * ec[(size_t)s * CE];
    if (t == 0) acc = 0.f;
    bf16 cb = (bf16)acc;
    xbf[(size_t)b * CDE + CD + e] = cb;
    ybf[(size_t)b * CDE + CD + e] = cb;
  }
  // teacher-forcing token: BOS(=1) at t==0 else targets[:, t]
  int tok = (t == 0) ? 1 : targets[b * CT + t];
  for (int d = tid; d < CD; d += 256)
    xbf[(size_t)b * CDE + d] = (bf16)emb[(size_t)tok * CD + d];
}

// ---------------------------------------------------------------------------
extern "C" void kernel_launch(void* const* d_in, const int* in_sizes, int n_in,
                              void* d_out, int out_size, void* d_ws, size_t ws_size,
                              hipStream_t stream) {
  (void)in_sizes; (void)n_in; (void)out_size; (void)ws_size;
  const float* enc    = (const float*)d_in[0];
  const int*   tgt    = (const int*)  d_in[1];
  const float* emb    = (const float*)d_in[2];
  const float* W_a    = (const float*)d_in[3];
  const float* b_Wa   = (const float*)d_in[4];
  const float* U_a    = (const float*)d_in[5];
  const float* b_Ua   = (const float*)d_in[6];
  const float* v_a    = (const float*)d_in[7];
  const float* b_va   = (const float*)d_in[8];
  const float* W_ih0  = (const float*)d_in[9];
  const float* W_hh0  = (const float*)d_in[10];
  const float* b_ih0  = (const float*)d_in[11];
  const float* b_hh0  = (const float*)d_in[12];
  const float* W_ih1  = (const float*)d_in[13];
  const float* W_hh1  = (const float*)d_in[14];
  const float* b_ih1  = (const float*)d_in[15];
  const float* b_hh1  = (const float*)d_in[16];
  const float* W_out  = (const float*)d_in[17];
  const float* b_out  = (const float*)d_in[18];
  float* out = (float*)d_out;

  // workspace carve-up (256B aligned)
  char* p = (char*)d_ws;
  auto take = [&](size_t bytes) {
    char* r = p;
    p += (bytes + 255) & ~(size_t)255;
    return (void*)r;
  };
  bf16* enc_bf  = (bf16*)take((size_t)CB * CS * CE * 2);
  bf16* wa_bf   = (bf16*)take((size_t)CA * CE * 2);
  bf16* wih0_bf = (bf16*)take((size_t)CG * CDE * 2);
  bf16* whh0_bf = (bf16*)take((size_t)CG * CD * 2);
  bf16* wih1_bf = (bf16*)take((size_t)CG * CD * 2);
  bf16* whh1_bf = (bf16*)take((size_t)CG * CD * 2);
  bf16* wout_bf = (bf16*)take((size_t)CV * CDE * 2);
  float* We     = (float*)take((size_t)CB * CS * CA * 4);
  float* c0     = (float*)take((size_t)CB * CD * 4);
  float* c1     = (float*)take((size_t)CB * CD * 4);
  float* h1     = (float*)take((size_t)CB * CD * 4);
  bf16* h0_bf   = (bf16*)take((size_t)CB * CD * 2);
  bf16* x_bf    = (bf16*)take((size_t)CB * CDE * 2);
  bf16* y_bf    = (bf16*)take((size_t)CB * CDE * 2);   // [h1 | ctx] bf16
  float* gates0 = (float*)take((size_t)CB * CG * 4);
  float* gates1 = (float*)take((size_t)CB * CG * 4);

  // state init (deterministic each call)
  zero_f32<<<(CB * CD + 255) / 256, 256, 0, stream>>>(c0, CB * CD);
  zero_f32<<<(CB * CD + 255) / 256, 256, 0, stream>>>(c1, CB * CD);
  zero_f32<<<(CB * CD + 255) / 256, 256, 0, stream>>>(h1, CB * CD);
  zero_bf<<<(CB * CD + 255) / 256, 256, 0, stream>>>(h0_bf, CB * CD);
  zero_bf<<<(CB * CDE + 255) / 256, 256, 0, stream>>>(y_bf, CB * CDE);

  // fp32 -> bf16 weight/activation conversions
  f2bf<<<4096, 256, 0, stream>>>(enc,   enc_bf,  (long long)CB * CS * CE);
  f2bf<<<256,  256, 0, stream>>>(W_a,   wa_bf,   (long long)CA * CE);
  f2bf<<<2048, 256, 0, stream>>>(W_ih0, wih0_bf, (long long)CG * CDE);
  f2bf<<<1024, 256, 0, stream>>>(W_hh0, whh0_bf, (long long)CG * CD);
  f2bf<<<1024, 256, 0, stream>>>(W_ih1, wih1_bf, (long long)CG * CD);
  f2bf<<<1024, 256, 0, stream>>>(W_hh1, whh1_bf, (long long)CG * CD);
  f2bf<<<8192, 256, 0, stream>>>(W_out, wout_bf, (long long)CV * CDE);

  // We = enc @ W_a^T + b_Wa : 2048 m-tiles x 8 n-tiles -> 16384 waves, 8/block
  we_gemm<<<2048, 256, 0, stream>>>(enc_bf, wa_bf, b_Wa, We);

  for (int t = 0; t < CT - 1; ++t) {
    attn_kernel<<<CB, 256, 0, stream>>>(enc, We, h1, U_a, b_Ua, v_a, b_va, emb,
                                        tgt, t, x_bf, y_bf);
    // layer 0: gates0 = x @ W_ih0^T + h0 @ W_hh0^T
    gates_gemm<<<32, 256, 0, stream>>>(x_bf, CDE, CDE, wih0_bf,
                                       h0_bf, CD, CD, whh0_bf, gates0);
    lstm_cell<<<64, 256, 0, stream>>>(gates0, b_ih0, b_hh0, c0, nullptr,
                                      h0_bf, CD);
    // layer 1: gates1 = h0_new @ W_ih1^T + h1_old @ W_hh1^T (h1_old in y_bf[:, :D])
    gates_gemm<<<32, 256, 0, stream>>>(h0_bf, CD, CD, wih1_bf,
                                       y_bf, CDE, CD, whh1_bf, gates1);
    lstm_cell<<<64, 256, 0, stream>>>(gates1, b_ih1, b_hh1, c1, h1,
                                      y_bf, CDE);
    // out[:, t, :] = [h1_new | ctx] @ W_out^T + b_out : 500 waves, 4/block
    out_gemm<<<125, 128, 0, stream>>>(y_bf, wout_bf, b_out, out, t);
  }
}
